// MultiHeadAttention_85925115724469
// MI455X (gfx1250) — matrix operand order repaired and checked by reading, not compiled, not run on hardware
//
#include <hip/hip_runtime.h>

// ---------------------------------------------------------------- types
typedef __attribute__((ext_vector_type(16))) unsigned short u16x16;
typedef __attribute__((ext_vector_type(16))) __bf16         bf16x16;
typedef __attribute__((ext_vector_type(8)))  float          f32x8;

static constexpr int Bn = 4, Sn = 2048, En = 1024, Hn = 16, DHn = 64;

// ---------------------------------------------------------------- helpers
static __device__ __forceinline__ unsigned short f2bf(float f) {
  unsigned u = __float_as_uint(f);
  unsigned r = u + 0x7FFFu + ((u >> 16) & 1u);   // round-to-nearest-even
  return (unsigned short)(r >> 16);
}

static __device__ __forceinline__ f32x8 wmma_bf16(u16x16 a, u16x16 b, f32x8 c) {
  union U { u16x16 u; bf16x16 b; };
  U ua; ua.u = a;
  U ub; ub.u = b;
  return __builtin_amdgcn_wmma_f32_16x16x32_bf16(false, ua.b, false, ub.b,
                                                 (short)0, c, false, false);
}

// A fragment (16x32 bf16) from row-major LDS tile. lane&15 = row, per ISA:
// VGPR i: K = (i<4 ? 2i : 16+2(i-4)) + 8*(lane>>4)  -> two contiguous 16B runs per lane
static __device__ __forceinline__ u16x16
load_a_frag(const unsigned short* base, int stride, int row0, int lane) {
  const unsigned short* p = base + (size_t)(row0 + (lane & 15)) * stride;
  int hi8 = 8 * (lane >> 4);
  u16x16 r;
#pragma unroll
  for (int i = 0; i < 8; ++i) {
    int k = ((i < 4) ? 2 * i : 16 + 2 * (i - 4)) + hi8;
    unsigned int wd = *(const unsigned int*)(p + k);
    r[2 * i]     = (unsigned short)(wd & 0xFFFFu);
    r[2 * i + 1] = (unsigned short)(wd >> 16);
  }
  return r;
}

// B fragment (32x16 bf16) from an n-major (transposed) LDS tile T[n][k]:
// lane&15 = column n, VGPR i holds K = (i<4 ? 2i : 16+2(i-4)) + 8*(lane>>4) + {0,1}  -> two 16B runs per lane
static __device__ __forceinline__ u16x16
load_bt_frag(const unsigned short* base, int stride, int col0, int koff, int lane) {
  const unsigned short* p =
      base + (size_t)(col0 + (lane & 15)) * stride + koff + 8 * (lane >> 4);
  u16x16 r;
#pragma unroll
  for (int i = 0; i < 16; ++i) r[i] = p[(i < 8) ? i : i + 8];
  return r;
}

// B fragment (32x16 bf16) gathered from a k-major LDS tile T[k][d] (for P*V)
static __device__ __forceinline__ u16x16
load_b_gather(const unsigned short* base, int stride, int col0, int lane) {
  int d  = col0 + (lane & 15);
  int kb = 8 * (lane >> 4);
  u16x16 r;
#pragma unroll
  for (int i = 0; i < 16; ++i) r[i] = base[(size_t)(kb + ((i < 8) ? i : i + 8)) * stride + d];
  return r;
}

// ---------------------------------------------------------------- kernel 0: conversions
__global__ __launch_bounds__(256) void cvt_f32_bf16(const float* __restrict__ src,
                                                    unsigned short* __restrict__ dst, int n) {
  int i = blockIdx.x * blockDim.x + threadIdx.x;
  if (i < n) dst[i] = f2bf(src[i]);
}

// pack wq/wk/wv [H,E,DH] fp32 -> wT[n][k] bf16, n = part*1024 + h*64 + d, k = e
__global__ __launch_bounds__(256) void cvt_wqkv(const float* __restrict__ wq,
                                                const float* __restrict__ wk,
                                                const float* __restrict__ wv,
                                                unsigned short* __restrict__ wT) {
  int idx = blockIdx.x * blockDim.x + threadIdx.x;  // 3072*1024
  int n = idx >> 10, k = idx & 1023;
  int part = n >> 10, rem = n & 1023;
  int h = rem >> 6, d = rem & 63;
  const float* w = (part == 0) ? wq : (part == 1) ? wk : wv;
  wT[idx] = f2bf(w[((size_t)h * En + k) * DHn + d]);
}

// ---------------------------------------------------------------- kernel 1: QKV GEMM
// C[8192,3072] = xb[8192,1024] x wT^T ; block tile 128x128, 8 waves x (32x64), 8 WMMA/iter
__global__ __launch_bounds__(256) void qkv_gemm(const unsigned short* __restrict__ xb,
                                                const unsigned short* __restrict__ wT,
                                                unsigned short* __restrict__ Qg,
                                                unsigned short* __restrict__ Kg,
                                                unsigned short* __restrict__ Vg) {
  __shared__ __align__(16) unsigned short As[128 * 40];   // rows padded to 80B
  __shared__ __align__(16) unsigned short Bs[128 * 40];
  const int t = threadIdx.x, lane = t & 31, w = t >> 5;
  const int ln = lane & 15, hi = lane >> 4;
  const int m0 = blockIdx.x * 128, n0 = blockIdx.y * 128;
  const int wm = (w >> 1) * 32, wn = (w & 1) * 64;
  f32x8 acc[2][4] = {};
#pragma unroll 1
  for (int kt = 0; kt < En; kt += 32) {
#pragma unroll
    for (int i = 0; i < 2; ++i) {
      int s = t + i * 256, r = s >> 2, c = (s & 3) * 8;
      *(uint4*)(As + r * 40 + c) = *(const uint4*)(xb + (size_t)(m0 + r) * En + kt + c);
      *(uint4*)(Bs + r * 40 + c) = *(const uint4*)(wT + (size_t)(n0 + r) * En + kt + c);
    }
    if (kt + 32 < En) {
      __builtin_prefetch(xb + (size_t)(m0 + (t >> 2)) * En + kt + 32 + (t & 3) * 8, 0, 1);
      __builtin_prefetch(wT + (size_t)(n0 + (t >> 2)) * En + kt + 32 + (t & 3) * 8, 0, 1);
    }
    __syncthreads();
    u16x16 a0 = load_a_frag(As, 40, wm, lane);
    u16x16 a1 = load_a_frag(As, 40, wm + 16, lane);
#pragma unroll
    for (int ni = 0; ni < 4; ++ni) {
      u16x16 b = load_bt_frag(Bs, 40, wn + ni * 16, 0, lane);
      acc[0][ni] = wmma_bf16(a0, b, acc[0][ni]);
      acc[1][ni] = wmma_bf16(a1, b, acc[1][ni]);
    }
    __syncthreads();
  }
#pragma unroll
  for (int mi = 0; mi < 2; ++mi)
#pragma unroll
    for (int ni = 0; ni < 4; ++ni) {
      int gcol = n0 + wn + ni * 16 + ln;
      int part = gcol >> 10, rem = gcol & 1023;
      int hh = rem >> 6, dd = rem & 63;
      unsigned short* dst = (part == 0) ? Qg : (part == 1) ? Kg : Vg;
#pragma unroll
      for (int e = 0; e < 8; ++e) {
        int grow = m0 + wm + mi * 16 + e + 8 * hi;
        int bb = grow >> 11, ss = grow & 2047;
        dst[((size_t)(bb * Hn + hh) * Sn + ss) * DHn + dd] = f2bf(acc[mi][ni][e]);
      }
    }
}

// ---------------------------------------------------------------- kernel 2: flash attention
// grid(S/128, B*H), 256 threads = 8 waves, each wave owns 16 q rows; K/V tile shared by 8 waves
__global__ __launch_bounds__(256) void attn_kernel(const unsigned short* __restrict__ Qg,
                                                   const unsigned short* __restrict__ Kg,
                                                   const unsigned short* __restrict__ Vg,
                                                   unsigned short* __restrict__ Yg) {
  __shared__ __align__(16) unsigned short Kl[32 * 72];      // 32 keys x 64 d (padded 144B)
  __shared__ __align__(16) unsigned short Vl[32 * 72];
  __shared__ __align__(16) unsigned short Pl[8][16 * 36];   // per-wave P scratch 16x32
  const int t = threadIdx.x, lane = t & 31, w = t >> 5;
  const int ln = lane & 15, hi = lane >> 4;
  const int q0 = blockIdx.x * 128;
  const int bh = blockIdx.y;
  const int b = bh >> 4, h = bh & 15;

  // Q fragments (d 0..31, 32..63) straight from global in A-layout
  const unsigned short* Qp = Qg + ((size_t)bh * Sn + q0 + w * 16) * DHn;
  u16x16 qf[2];
#pragma unroll
  for (int f = 0; f < 2; ++f)
#pragma unroll
    for (int i = 0; i < 8; ++i) {
      int k = f * 32 + ((i < 4) ? 2 * i : 16 + 2 * (i - 4)) + 8 * hi;
      unsigned int wd = *(const unsigned int*)(Qp + (size_t)ln * DHn + k);
      qf[f][2 * i]     = (unsigned short)(wd & 0xFFFFu);
      qf[f][2 * i + 1] = (unsigned short)(wd >> 16);
    }

  f32x8 acc[4] = {};
  float m[8], l[8];
#pragma unroll
  for (int e = 0; e < 8; ++e) { m[e] = -1e30f; l[e] = 0.f; }

  const unsigned short* Kbase = Kg + (size_t)bh * Sn * DHn;
  const unsigned short* Vbase = Vg + (size_t)bh * Sn * DHn;

#pragma unroll 1
  for (int kt0 = 0; kt0 < Sn; kt0 += 32) {
    {  // cooperative tile loads: 256 threads x one 16B chunk per tile (32 rows x 8 chunks)
      int r = t >> 3, c = (t & 7) * 8;
      *(uint4*)(Kl + r * 72 + c) = *(const uint4*)(Kbase + (size_t)(kt0 + r) * DHn + c);
      *(uint4*)(Vl + r * 72 + c) = *(const uint4*)(Vbase + (size_t)(kt0 + r) * DHn + c);
    }
    if (kt0 + 32 < Sn)
      __builtin_prefetch(Kbase + (size_t)(kt0 + 32 + (t >> 3)) * DHn + (t & 7) * 8, 0, 1);
    __syncthreads();

    // scores: Q(16x64) x K^T(64x32) -> two 16x16 C tiles
    f32x8 s0 = {}, s1 = {};
    {
      u16x16 kb;
      kb = load_bt_frag(Kl, 72, 0, 0, lane);   s0 = wmma_bf16(qf[0], kb, s0);
      kb = load_bt_frag(Kl, 72, 0, 32, lane);  s0 = wmma_bf16(qf[1], kb, s0);
      kb = load_bt_frag(Kl, 72, 16, 0, lane);  s1 = wmma_bf16(qf[0], kb, s1);
      kb = load_bt_frag(Kl, 72, 16, 32, lane); s1 = wmma_bf16(qf[1], kb, s1);
    }

    // online softmax (row groups of 16 lanes; xor masks stay inside the group)
    float sc[8];
#pragma unroll
    for (int e = 0; e < 8; ++e) {
      float a0 = s0[e] * 0.125f, a1 = s1[e] * 0.125f;   // 1/sqrt(64)
      s0[e] = a0; s1[e] = a1;
      float r = fmaxf(a0, a1);
      r = fmaxf(r, __shfl_xor(r, 1));
      r = fmaxf(r, __shfl_xor(r, 2));
      r = fmaxf(r, __shfl_xor(r, 4));
      r = fmaxf(r, __shfl_xor(r, 8));
      float mn = fmaxf(m[e], r);
      sc[e] = __expf(m[e] - mn);
      m[e] = mn;
      l[e] *= sc[e];
    }
#pragma unroll
    for (int c = 0; c < 4; ++c)
#pragma unroll
      for (int e = 0; e < 8; ++e) acc[c][e] *= sc[e];

    // P = exp(s-m): C-layout -> per-wave LDS -> A-layout
    unsigned short* Pp = &Pl[w][0];
#pragma unroll
    for (int e = 0; e < 8; ++e) {
      float p0 = __expf(s0[e] - m[e]);
      float p1 = __expf(s1[e] - m[e]);
      float r = p0 + p1;
      r += __shfl_xor(r, 1);
      r += __shfl_xor(r, 2);
      r += __shfl_xor(r, 4);
      r += __shfl_xor(r, 8);
      l[e] += r;
      Pp[(e + 8 * hi) * 36 + ln]      = f2bf(p0);
      Pp[(e + 8 * hi) * 36 + 16 + ln] = f2bf(p1);
    }

    u16x16 pf = load_a_frag(Pp, 36, 0, lane);
#pragma unroll
    for (int c = 0; c < 4; ++c) {
      u16x16 vf = load_b_gather(Vl, 72, c * 16, lane);
      acc[c] = wmma_bf16(pf, vf, acc[c]);
    }
    __syncthreads();
  }

  // normalize and write y[B,S,E] bf16 (head-major cols)
#pragma unroll
  for (int c = 0; c < 4; ++c) {
    int gcol = h * DHn + c * 16 + ln;
#pragma unroll
    for (int e = 0; e < 8; ++e) {
      int grow = b * Sn + q0 + w * 16 + e + 8 * hi;
      Yg[(size_t)grow * En + gcol] = f2bf(acc[c][e] / l[e]);
    }
  }
}

// ---------------------------------------------------------------- kernel 3: output projection
// out[8192,1024] = y x wo^T + bo ; block tile 128x128, 8 waves x (32x64)
__global__ __launch_bounds__(256) void out_gemm(const unsigned short* __restrict__ yb,
                                                const unsigned short* __restrict__ wob,
                                                const float* __restrict__ bo,
                                                float* __restrict__ out) {
  __shared__ __align__(16) unsigned short As[128 * 40];
  __shared__ __align__(16) unsigned short Bs[128 * 40];
  const int t = threadIdx.x, lane = t & 31, w = t >> 5;
  const int ln = lane & 15, hi = lane >> 4;
  const int m0 = blockIdx.x * 128, n0 = blockIdx.y * 128;
  const int wm = (w >> 1) * 32, wn = (w & 1) * 64;
  f32x8 acc[2][4] = {};
#pragma unroll 1
  for (int kt = 0; kt < En; kt += 32) {
#pragma unroll
    for (int i = 0; i < 2; ++i) {
      int s = t + i * 256, r = s >> 2, c = (s & 3) * 8;
      *(uint4*)(As + r * 40 + c) = *(const uint4*)(yb + (size_t)(m0 + r) * En + kt + c);
      *(uint4*)(Bs + r * 40 + c) = *(const uint4*)(wob + (size_t)(n0 + r) * En + kt + c);
    }
    if (kt + 32 < En) {
      __builtin_prefetch(yb + (size_t)(m0 + (t >> 2)) * En + kt + 32 + (t & 3) * 8, 0, 1);
      __builtin_prefetch(wob + (size_t)(n0 + (t >> 2)) * En + kt + 32 + (t & 3) * 8, 0, 1);
    }
    __syncthreads();
    u16x16 a0 = load_a_frag(As, 40, wm, lane);
    u16x16 a1 = load_a_frag(As, 40, wm + 16, lane);
#pragma unroll
    for (int ni = 0; ni < 4; ++ni) {
      u16x16 b = load_bt_frag(Bs, 40, wn + ni * 16, 0, lane);
      acc[0][ni] = wmma_bf16(a0, b, acc[0][ni]);
      acc[1][ni] = wmma_bf16(a1, b, acc[1][ni]);
    }
    __syncthreads();
  }
#pragma unroll
  for (int mi = 0; mi < 2; ++mi)
#pragma unroll
    for (int ni = 0; ni < 4; ++ni) {
      int gcol = n0 + wn + ni * 16 + ln;
      float bias = bo[gcol];
#pragma unroll
      for (int e = 0; e < 8; ++e) {
        int grow = m0 + wm + mi * 16 + e + 8 * hi;
        out[(size_t)grow * En + gcol] = acc[mi][ni][e] + bias;
      }
    }
}

// ---------------------------------------------------------------- launch
extern "C" void kernel_launch(void* const* d_in, const int* in_sizes, int n_in,
                              void* d_out, int out_size, void* d_ws, size_t ws_size,
                              hipStream_t stream) {
  const float* x  = (const float*)d_in[0];
  const float* wq = (const float*)d_in[1];
  const float* wk = (const float*)d_in[2];
  const float* wv = (const float*)d_in[3];
  const float* wo = (const float*)d_in[4];
  const float* bo = (const float*)d_in[5];
  float* out = (float*)d_out;

  char* ws = (char*)d_ws;
  size_t off = 0;
  auto take = [&](size_t bytes) -> char* {
    char* p = ws + off;
    off += (bytes + 255) & ~(size_t)255;
    return p;
  };
  unsigned short* xb  = (unsigned short*)take((size_t)Bn * Sn * En * 2);        // 16 MB
  unsigned short* wT  = (unsigned short*)take((size_t)3 * En * En * 2);         // 6 MB
  unsigned short* wob = (unsigned short*)take((size_t)En * En * 2);             // 2 MB
  unsigned short* Qg  = (unsigned short*)take((size_t)Bn * Hn * Sn * DHn * 2);  // 16 MB
  unsigned short* Kg  = (unsigned short*)take((size_t)Bn * Hn * Sn * DHn * 2);
  unsigned short* Vg  = (unsigned short*)take((size_t)Bn * Hn * Sn * DHn * 2);
  unsigned short* Yg  = (unsigned short*)take((size_t)Bn * Sn * En * 2);        // 16 MB

  const int nX = Bn * Sn * En;  // 8388608
  cvt_f32_bf16<<<nX / 256, 256, 0, stream>>>(x, xb, nX);
  cvt_wqkv<<<(3 * En * En) / 256, 256, 0, stream>>>(wq, wk, wv, wT);
  cvt_f32_bf16<<<(En * En) / 256, 256, 0, stream>>>(wo, wob, En * En);

  qkv_gemm<<<dim3((Bn * Sn) / 128, (3 * En) / 128), 256, 0, stream>>>(xb, wT, Qg, Kg, Vg);
  attn_kernel<<<dim3(Sn / 128, Bn * Hn), 256, 0, stream>>>(Qg, Kg, Vg, Yg);
  out_gemm<<<dim3((Bn * Sn) / 128, En / 128), 256, 0, stream>>>(Yg, wob, bo, out);
}
